// CKAN_63754494542357
// MI455X (gfx1250) — matrix operand order repaired and checked by reading, not compiled, not run on hardware
//
#include <hip/hip_runtime.h>
#include <hip/hip_bf16.h>
#include <stdint.h>

typedef _Float16 half16 __attribute__((ext_vector_type(16)));
typedef _Float16 half8  __attribute__((ext_vector_type(8)));
typedef _Float16 half4  __attribute__((ext_vector_type(4)));
typedef float    f32x8  __attribute__((ext_vector_type(8)));
typedef int      i32x4  __attribute__((vector_size(16)));   // matches builtin pointee type

#define BATCH    4096
#define KNBR     64
#define DIM      64
#define N_ENTITY 200000

union hvec { half16 v; half8 h[2]; };

// ---- CDNA5 async global->LDS path (ASYNCcnt-tracked), with safe fallback ----
#if defined(__has_builtin)
#  if __has_builtin(__builtin_amdgcn_global_load_async_to_lds_b128) && \
      __has_builtin(__builtin_amdgcn_s_wait_asynccnt)
#    define USE_ASYNC 1
#  endif
#endif
#ifndef USE_ASYNC
#  define USE_ASYNC 0
#endif

#if USE_ASYNC
// as(3) pointers are 32-bit LDS offsets == low 32 bits of the generic address;
// as(1) pointers equal the generic 64-bit address (ISA 10.2).
#define AS1P(p) ((__attribute__((address_space(1))) i32x4*)(unsigned long long)(uintptr_t)(p))
#define AS3P(p) ((__attribute__((address_space(3))) i32x4*)(unsigned int)(uintptr_t)(p))
#define ASYNC_B128(gsrc, ldst) \
    __builtin_amdgcn_global_load_async_to_lds_b128(AS1P(gsrc), AS3P(ldst), 0, 0)
#endif

// --- one streaming pass: entity/relation/W1/W2 f32 -> f16 tables in workspace ---
__global__ void ckan_to_f16(const float* __restrict__ entity, const float* __restrict__ relation,
                            const float* __restrict__ W1, const float* __restrict__ W2,
                            _Float16* __restrict__ ent_h, _Float16* __restrict__ rel_h,
                            _Float16* __restrict__ W1h, _Float16* __restrict__ W2h) {
    const int  gid = blockIdx.x * blockDim.x + threadIdx.x;
    const int  nth = gridDim.x * blockDim.x;
    const long n4  = (long)N_ENTITY * DIM / 4;
    const float4* e4 = (const float4*)entity;
    for (long i = gid; i < n4; i += nth) {
        float4 v = e4[i];
        half4  h = {(_Float16)v.x, (_Float16)v.y, (_Float16)v.z, (_Float16)v.w};
        *(half4*)(ent_h + i * 4) = h;
    }
    if (gid < 2048)                    rel_h[gid]              = (_Float16)relation[gid];
    else if (gid < 2048 + 8192)        W1h[gid - 2048]         = (_Float16)W1[gid - 2048];
    else if (gid < 2048 + 8192 + 4096) W2h[gid - 2048 - 8192]  = (_Float16)W2[gid - 2048 - 8192];
}

// --- per-(side,layer,b) attention block: async-staged WMMA MLP + softmax + t-sum ---
__global__ __launch_bounds__(128)
void ckan_attn(const _Float16* __restrict__ ent_h, const _Float16* __restrict__ rel_h,
               const _Float16* __restrict__ W1h, const _Float16* __restrict__ W2h,
               const float* __restrict__ W3,
               const int* __restrict__ uh, const int* __restrict__ ur, const int* __restrict__ ut,
               const int* __restrict__ ih, const int* __restrict__ ir, const int* __restrict__ it,
               float* __restrict__ att_out) {
    __shared__ _Float16 Ax[64 * 128];   // [edge][h(64)|r(64)]  16 KB
    __shared__ _Float16 X1[64 * 64];    // layer-1 activations   8 KB
    __shared__ _Float16 W1s[64 * 128];  // 16 KB
    __shared__ _Float16 W2s[64 * 64];   //  8 KB
    __shared__ float    W3s[64];
    __shared__ float    part[4][16][16];
    __shared__ float    attv[64];
    __shared__ float    wv[64];
    __shared__ float    red[128];

    const int tid   = threadIdx.x;
    const int blk   = blockIdx.x;
    const int b     = blk & (BATCH - 1);
    const int sl    = blk >> 12;          // side*2 + layer
    const int layer = sl & 1;
    const int side  = sl >> 1;

    const int* hidx = side ? ih : uh;
    const int* ridx = side ? ir : ur;
    const int* tidx = side ? it : ut;
    const long base = ((long)layer * BATCH + b) * KNBR;

    // ---- stage weights + gathered A-tile into LDS ----
#if USE_ASYNC
    for (int c = tid; c < 1024; c += 128)   // W1s: 16 KB in b128 chunks
        ASYNC_B128((const char*)W1h + c * 16, (char*)W1s + c * 16);
    for (int c = tid; c < 512; c += 128)    // W2s: 8 KB
        ASYNC_B128((const char*)W2h + c * 16, (char*)W2s + c * 16);
#else
    for (int c = tid; c < 1024; c += 128)
        *(half8*)((char*)W1s + c * 16) = *(const half8*)((const char*)W1h + c * 16);
    for (int c = tid; c < 512; c += 128)
        *(half8*)((char*)W2s + c * 16) = *(const half8*)((const char*)W2h + c * 16);
#endif
    if (tid < 64) W3s[tid] = W3[tid];

    {   // thread (2k+w): edge k; w==0 -> head emb (cols 0..63), w==1 -> relation (64..127)
        const int k = tid >> 1, w = tid & 1;
        const _Float16* src = w ? (rel_h + (long)ridx[base + k] * DIM)
                                : (ent_h + (long)hidx[base + k] * DIM);
        _Float16* dst = Ax + k * 128 + w * 64;
#if USE_ASYNC
        #pragma unroll
        for (int j = 0; j < 8; ++j) ASYNC_B128(src + j * 8, dst + j * 8);
        __builtin_amdgcn_s_wait_asynccnt(0);
#else
        #pragma unroll
        for (int j = 0; j < 8; ++j) *(half8*)(dst + j * 8) = *(const half8*)(src + j * 8);
#endif
    }
    __syncthreads();

    const int wave    = tid >> 5;
    const int lane    = tid & 31;
    const int lane16  = lane & 15;
    const int laneHi  = lane >> 4;
    const int rowBase = wave * 16;
    const int myRow   = rowBase + lane16;

    // ---- layer 1: [16x128] @ W1^T -> [16x64], relu, pack f16 to X1 ----
    hvec a1[4];
    #pragma unroll
    for (int t = 0; t < 4; ++t) {          // A frag per ISA 16-bit 16x32 layout
        const _Float16* p = Ax + myRow * 128 + 32 * t + 8 * laneHi;
        a1[t].h[0] = *(const half8*)(p);
        a1[t].h[1] = *(const half8*)(p + 16);
    }
    #pragma unroll
    for (int n = 0; n < 4; ++n) {
        f32x8 c = {};
        #pragma unroll
        for (int t = 0; t < 4; ++t) {      // B frag: row N of W1, K halves in ISA table order
            hvec bf;
            const _Float16* p = W1s + (n * 16 + lane16) * 128 + 32 * t + 8 * laneHi;
            bf.h[0] = *(const half8*)(p);
            bf.h[1] = *(const half8*)(p + 16);
            c = __builtin_amdgcn_wmma_f32_16x16x32_f16(false, a1[t].v, false, bf.v,
                                                       (short)0, c, false, false);
        }
        #pragma unroll
        for (int r = 0; r < 8; ++r) {      // C frag: row r+8*laneHi, col n*16+lane16
            float v = c[r] > 0.f ? c[r] : 0.f;
            X1[(rowBase + r + 8 * laneHi) * 64 + n * 16 + lane16] = (_Float16)v;
        }
    }
    __syncthreads();

    // ---- layer 2: [16x64] @ W2^T -> relu -> partial dot with W3 ----
    float pacc[8];
    #pragma unroll
    for (int r = 0; r < 8; ++r) pacc[r] = 0.f;
    hvec a2[2];
    #pragma unroll
    for (int t = 0; t < 2; ++t) {
        const _Float16* q = X1 + myRow * 64 + 32 * t + 8 * laneHi;
        a2[t].h[0] = *(const half8*)(q);
        a2[t].h[1] = *(const half8*)(q + 16);
    }
    #pragma unroll
    for (int n = 0; n < 4; ++n) {
        f32x8 c = {};
        #pragma unroll
        for (int t = 0; t < 2; ++t) {
            hvec bf;
            const _Float16* q = W2s + (n * 16 + lane16) * 64 + 32 * t + 8 * laneHi;
            bf.h[0] = *(const half8*)(q);
            bf.h[1] = *(const half8*)(q + 16);
            c = __builtin_amdgcn_wmma_f32_16x16x32_f16(false, a2[t].v, false, bf.v,
                                                       (short)0, c, false, false);
        }
        const float w3c = W3s[n * 16 + lane16];
        #pragma unroll
        for (int r = 0; r < 8; ++r) {
            float v = c[r] > 0.f ? c[r] : 0.f;
            pacc[r] += v * w3c;
        }
    }
    #pragma unroll
    for (int r = 0; r < 8; ++r) part[wave][r + 8 * laneHi][lane16] = pacc[r];
    __syncthreads();

    // ---- attention scalar: reduce 16 lane-partials, sigmoid ----
    if (tid < 64) {
        float s = 0.f;
        #pragma unroll
        for (int j = 0; j < 16; ++j) s += part[tid >> 4][tid & 15][j];
        attv[tid] = 1.f / (1.f + __expf(-s));
    }
    __syncthreads();

    // ---- softmax over K=64 (redundant per-thread scan; deterministic) ----
    float mx = -1e30f;
    for (int k = 0; k < 64; ++k) mx = fmaxf(mx, attv[k]);
    float sum = 0.f;
    for (int k = 0; k < 64; ++k) sum += __expf(attv[k] - mx);
    if (tid < 64) wv[tid] = __expf(attv[tid] - mx) / sum;
    __syncthreads();

    // ---- weighted sum of t_emb (f16 table): out[d] = sum_k w_k * t_emb[k][d] ----
    {
        const int d = tid & 63, hh = tid >> 6;
        float acc = 0.f;
        for (int k = hh * 32; k < hh * 32 + 32; ++k) {
            const long ti = tidx[base + k];
            acc += wv[k] * (float)ent_h[ti * DIM + d];
        }
        red[tid] = acc;
    }
    __syncthreads();
    if (tid < 64) {
        att_out[(((long)side * 2 + layer) * BATCH + b) * DIM + tid] = red[tid] + red[tid + 64];
    }
}

// --- combine: e_u, e_v, dot, sigmoid ---
__global__ __launch_bounds__(64)
void ckan_combine(const _Float16* __restrict__ ent_h, const int* __restrict__ items,
                  const int* __restrict__ uh, const int* __restrict__ ih,
                  const float* __restrict__ att_out, float* __restrict__ out) {
    __shared__ float red[64];
    const int b = blockIdx.x, d = threadIdx.x;
    float eu = att_out[((long)(0 * 2 + 0) * BATCH + b) * DIM + d]
             + att_out[((long)(0 * 2 + 1) * BATCH + b) * DIM + d];
    float ev = att_out[((long)(1 * 2 + 0) * BATCH + b) * DIM + d]
             + att_out[((long)(1 * 2 + 1) * BATCH + b) * DIM + d];
    const int* uh0 = uh + (long)b * KNBR;   // layer-0 slice
    const int* ih0 = ih + (long)b * KNBR;
    float mu = 0.f, mi = 0.f;
    for (int k = 0; k < KNBR; ++k) {
        mu += (float)ent_h[(long)uh0[k] * DIM + d];
        mi += (float)ent_h[(long)ih0[k] * DIM + d];
    }
    eu += mu * (1.f / 64.f);
    ev += mi * (1.f / 64.f) + (float)ent_h[(long)items[b] * DIM + d];
    red[d] = eu * ev;
    __syncthreads();
    for (int s = 32; s > 0; s >>= 1) {
        if (d < s) red[d] += red[d + s];
        __syncthreads();
    }
    if (d == 0) out[b] = 1.f / (1.f + __expf(-red[0]));
}

extern "C" void kernel_launch(void* const* d_in, const int* in_sizes, int n_in,
                              void* d_out, int out_size, void* d_ws, size_t ws_size,
                              hipStream_t stream) {
    const float* entity   = (const float*)d_in[0];
    const float* relation = (const float*)d_in[1];
    const float* W1 = (const float*)d_in[2];
    const float* W2 = (const float*)d_in[3];
    const float* W3 = (const float*)d_in[4];
    const int* items = (const int*)d_in[5];
    const int* uh = (const int*)d_in[6];
    const int* ur = (const int*)d_in[7];
    const int* ut = (const int*)d_in[8];
    const int* ih = (const int*)d_in[9];
    const int* ir = (const int*)d_in[10];
    const int* it = (const int*)d_in[11];

    // workspace: [att_out 8 MiB][W1h 16K][W2h 8K][rel_h 4K][ent_h 25.6 MB]
    float*    att_out = (float*)d_ws;
    char*     wsb  = (char*)d_ws + (size_t)8 * 1024 * 1024;
    _Float16* W1h   = (_Float16*)(wsb);
    _Float16* W2h   = (_Float16*)(wsb + 16384);
    _Float16* rel_h = (_Float16*)(wsb + 24576);
    _Float16* ent_h = (_Float16*)(wsb + 28672);

    ckan_to_f16<<<2048, 256, 0, stream>>>(entity, relation, W1, W2, ent_h, rel_h, W1h, W2h);
    ckan_attn<<<4 * BATCH, 128, 0, stream>>>(ent_h, rel_h, W1h, W2h, W3,
                                             uh, ur, ut, ih, ir, it, att_out);
    ckan_combine<<<BATCH, 64, 0, stream>>>(ent_h, items, uh, ih, att_out, (float*)d_out);
}
